// Attention_50775103373364
// MI455X (gfx1250) — compile-verified, matrix operand order checked by reading, not run on hardware
//
#include <hip/hip_runtime.h>

// ---------------------------------------------------------------------------
// MI455X (gfx1250) fused ViT causal attention, bf16 WMMA pipeline.
//   prep: X[f32]->Xb[bf16], W*->Wt*[bf16, transposed]
//   GEMM (async double-buffered LDS staging): Q,K,V[bf16]
//   flash-attn (K staged via async-to-LDS): ctx[bf16]
//   GEMM: out[f32]
// All matmuls: v_wmma_f32_16x16x32_bf16 (wave32, f32 accum).
// ---------------------------------------------------------------------------

typedef __attribute__((ext_vector_type(16))) __bf16 v16bf;
typedef __attribute__((ext_vector_type(8)))  __bf16 v8bf;
typedef __attribute__((ext_vector_type(4)))  __bf16 v4bf;
typedef __attribute__((ext_vector_type(8)))  float  v8f;
typedef int b128_t __attribute__((__vector_size__(16)));   // 4 x i32

#define B_  256
#define T_  197
#define C_  768
#define H_  12
#define D_  64
#define M_  (B_*T_)   // 50432 = 394 * 128
#define W2_ (C_*C_)   // 589824

// ---- CDNA5 async global->LDS copies (guarded: fall back to manual staging) --
#if defined(__has_builtin)
# if __has_builtin(__builtin_amdgcn_global_load_async_to_lds_b128)
#  define USE_ASYNC 1
# endif
#endif
#ifndef USE_ASYNC
# define USE_ASYNC 0
#endif

#if USE_ASYNC
# if __has_builtin(__builtin_amdgcn_s_wait_asynccnt)
#  define WAIT_ASYNC() __builtin_amdgcn_s_wait_asynccnt(0)
# else
#  define WAIT_ASYNC() asm volatile("s_wait_asynccnt 0" ::: "memory")
# endif
// 16-byte async copy: ASYNCcnt-tracked, no VGPR data path.
// Param 0: AS1 (global) b128*, param 1: AS3 (LDS) b128*, then imm offset, cpol.
__device__ __forceinline__ void async_cp16(const void* g, void* l) {
  __builtin_amdgcn_global_load_async_to_lds_b128(
      (__attribute__((address_space(1))) b128_t*)(unsigned long long)g,
      (__attribute__((address_space(3))) b128_t*)(unsigned int)(unsigned long long)l,
      0, 0);
}
#endif

// Load a 16x32 bf16 A/B fragment: per-lane two 16B chunks at p[0..7], p[16..23]
// (CDNA5 WMMA 16-bit operand layout; the +8 for hi lanes is baked into p).
__device__ __forceinline__ v16bf ld_frag(const __bf16* p) {
  v8bf lo = *(const v8bf*)(p);
  v8bf hi = *(const v8bf*)(p + 16);
  return __builtin_shufflevector(lo, hi, 0,1,2,3,4,5,6,7,8,9,10,11,12,13,14,15);
}

__device__ __forceinline__ v8f wmma_bf16(v16bf a, v16bf b, v8f c) {
  return __builtin_amdgcn_wmma_f32_16x16x32_bf16(false, a, false, b,
                                                 (short)0, c, false, false);
}

// ---------------------------------------------------------------------------
// Prep: fp32 -> bf16 convert (X), and convert+transpose weights Wt[n][k].
// ---------------------------------------------------------------------------
__global__ __launch_bounds__(256)
void cvt_kernel(const float* __restrict__ X, __bf16* __restrict__ Y) {
  const size_t i = ((size_t)blockIdx.x * 256 + threadIdx.x) * 4;
  float4 v = *(const float4*)(X + i);
  v4bf t;
  t.x = (__bf16)v.x; t.y = (__bf16)v.y; t.z = (__bf16)v.z; t.w = (__bf16)v.w;
  *(v4bf*)(Y + i) = t;
}

__global__ __launch_bounds__(256)
void wtr_kernel(const float* __restrict__ W, __bf16* __restrict__ Wt) {
  const int idx = blockIdx.x * 256 + threadIdx.x;  // 589824 = 2304 * 256
  const int k = idx / C_, n = idx % C_;
  Wt[(size_t)n * C_ + k] = (__bf16)W[idx];
}

// ---------------------------------------------------------------------------
// GEMM: out[M x 768] = A[M x 768](bf16) @ Wt^T + bias, Wt is [n][k] bf16.
// Block tile 128x128, BK=32, 8 waves -> each wave 32x64 (2x4 WMMA tiles).
// Async path: double-buffered LDS tiles, one barrier per K-step.
// ---------------------------------------------------------------------------
template<bool OBF16>
__global__ __launch_bounds__(256)
void gemm_kernel(const __bf16* __restrict__ A, const __bf16* __restrict__ Bt,
                 const float* __restrict__ bias, void* __restrict__ Op) {
#if USE_ASYNC
  constexpr int NBUF = 2;
#else
  constexpr int NBUF = 1;
#endif
  __shared__ __align__(16) __bf16 Als[NBUF][128*40];
  __shared__ __align__(16) __bf16 Bls[NBUF][128*40];
  const int tid  = threadIdx.x;
  const int lane = tid & 31, wid = tid >> 5;
  const int l16  = lane & 15, hi = lane >> 4;
  const int wm   = wid & 3,  wn = wid >> 2;
  const size_t m0 = (size_t)blockIdx.x * 128;
  const size_t n0 = (size_t)blockIdx.y * 128;

  v8f acc[2][4] = {};

  auto compute = [&](int buf) {
    v16bf af[2], bfr[4];
#pragma unroll
    for (int mi = 0; mi < 2; ++mi)
      af[mi]  = ld_frag(&Als[buf][(wm*32 + mi*16 + l16)*40 + hi*8]);
#pragma unroll
    for (int ni = 0; ni < 4; ++ni)
      bfr[ni] = ld_frag(&Bls[buf][(wn*64 + ni*16 + l16)*40 + hi*8]);
#pragma unroll
    for (int mi = 0; mi < 2; ++mi)
#pragma unroll
      for (int ni = 0; ni < 4; ++ni)
        acc[mi][ni] = wmma_bf16(af[mi], bfr[ni], acc[mi][ni]);
  };

#if USE_ASYNC
  auto stage = [&](int kt, int buf) {
#pragma unroll
    for (int i = 0; i < 2; ++i) {                    // 512 16B chunks / matrix
      const int lin = tid + i * 256;
      const int row = lin >> 2, c8 = (lin & 3) << 3;
      async_cp16(A  + (m0 + row) * C_ + kt + c8, &Als[buf][row*40 + c8]);
      async_cp16(Bt + (n0 + row) * C_ + kt + c8, &Bls[buf][row*40 + c8]);
    }
  };
  stage(0, 0);
  for (int it = 0; it < C_/32; ++it) {
    WAIT_ASYNC();            // this wave's current tile has landed
    __syncthreads();         // everyone's tile landed; prev compute finished
    if (it + 1 < C_/32) stage((it + 1) * 32, (it + 1) & 1);  // overlap
    compute(it & 1);
  }
#else
  for (int kt = 0; kt < C_; kt += 32) {
    __syncthreads();
#pragma unroll
    for (int i = 0; i < 2; ++i) {
      const int lin = tid + i * 256;
      const int row = lin >> 2, c8 = (lin & 3) << 3;
      *(v8bf*)&Als[0][row*40 + c8] = *(const v8bf*)(A  + (m0 + row)*C_ + kt + c8);
      *(v8bf*)&Bls[0][row*40 + c8] = *(const v8bf*)(Bt + (n0 + row)*C_ + kt + c8);
    }
    __syncthreads();
    compute(0);
  }
#endif

  // Epilogue: C/D layout -> lane l16 = column, VGPR r = row (+8 for hi lanes)
#pragma unroll
  for (int ni = 0; ni < 4; ++ni) {
    const size_t gc = n0 + wn*64 + ni*16 + l16;
    const float bv = bias[gc];
#pragma unroll
    for (int mi = 0; mi < 2; ++mi) {
#pragma unroll
      for (int r = 0; r < 8; ++r) {
        const size_t gr = m0 + wm*32 + mi*16 + r + hi*8;
        const float v = acc[mi][ni][r] + bv;
        if constexpr (OBF16) ((__bf16*)Op)[gr*C_ + gc] = (__bf16)v;
        else                 ((float*)Op)[gr*C_ + gc]  = v;
      }
    }
  }
}

// ---------------------------------------------------------------------------
// Flash attention per (b, h, 128-query slab). K (208x64, stride-72 pad,
// async-staged) and V^T (64x208, manual transpose, zero-padded) in LDS.
// Each wave owns 16 query rows; per 32-key block: 4 WMMAs for S, online
// softmax (xor-shuffle row reduce), P repack via per-wave LDS, 4 WMMAs PV.
// ---------------------------------------------------------------------------
#define KST 72
#define VST 208

__global__ __launch_bounds__(256)
void flash_kernel(const __bf16* __restrict__ Q, const __bf16* __restrict__ Kg,
                  const __bf16* __restrict__ V, __bf16* __restrict__ O) {
  __shared__ __align__(16) __bf16 Kl[208*KST];  // [key][d]
  __shared__ __align__(16) __bf16 Vl[64*VST];   // [d][key] (transposed)
  __shared__ __align__(16) __bf16 Pl[8*512];    // per-wave P repack buffer
  const int tid  = threadIdx.x;
  const int lane = tid & 31, wid = tid >> 5;
  const int l16  = lane & 15, hi = lane >> 4;
  const int b = blockIdx.z, h = blockIdx.y;
  const size_t base = (size_t)b * T_ * C_ + h * D_;

#if USE_ASYNC
  // K rows 197..207 read adjacent (valid) ws memory; those keys are masked.
  for (int idx = tid; idx < 208*8; idx += 256) {
    const int t = idx >> 3, c8 = (idx & 7) << 3;
    async_cp16(Kg + base + (size_t)t * C_ + c8, &Kl[t*KST + c8]);
  }
#endif
  for (int idx = tid; idx < 208*64; idx += 256) {
    const int t = idx >> 6, d = idx & 63;
    __bf16 vv = (__bf16)0.f;
    if (t < T_) vv = V[base + (size_t)t * C_ + d];
    Vl[d*VST + t] = vv;                     // zero pad: keeps PV NaN-free
#if !USE_ASYNC
    __bf16 kv = (__bf16)0.f;
    if (t < T_) kv = Kg[base + (size_t)t * C_ + d];
    Kl[t*KST + d] = kv;
#endif
  }
  for (int idx = tid; idx < 8*512; idx += 256) Pl[idx] = (__bf16)0.f;
#if USE_ASYNC
  WAIT_ASYNC();
#endif
  __syncthreads();

  const int q0 = blockIdx.x * 128 + wid * 16;
  if (q0 >= T_) return;  // uniform per-wave, after the only barrier

  v16bf qf[2];
#pragma unroll
  for (int c = 0; c < 2; ++c) {
    const int q = q0 + l16;
    if (q < T_) qf[c] = ld_frag(Q + base + (size_t)q * C_ + c*32 + hi*8);
    else        { v16bf z = {}; qf[c] = z; }
  }

  float m_[8], l_[8], sf_[8];
#pragma unroll
  for (int r = 0; r < 8; ++r) { m_[r] = -3.0e38f; l_[r] = 0.f; }
  v8f acc[4] = {};

  __bf16* Pw = &Pl[wid * 512];
  const int kend = (q0 + 16 < T_) ? (q0 + 16) : T_;   // causal upper bound

  for (int kb = 0; kb < kend; kb += 32) {
    v8f S0 = {}, S1 = {};
#pragma unroll
    for (int c = 0; c < 2; ++c) {
      v16bf k0 = ld_frag(&Kl[(kb      + l16)*KST + c*32 + hi*8]);
      v16bf k1 = ld_frag(&Kl[(kb + 16 + l16)*KST + c*32 + hi*8]);
      S0 = wmma_bf16(qf[c], k0, S0);
      S1 = wmma_bf16(qf[c], k1, S1);
    }
    const int key0 = kb + l16, key1 = kb + 16 + l16;
#pragma unroll
    for (int r = 0; r < 8; ++r) {
      const int qr = q0 + r + hi*8;
      float s0 = S0[r] * 0.125f;     // 1/sqrt(64)
      float s1 = S1[r] * 0.125f;
      if (key0 > qr || key0 >= T_) s0 = -3.0e38f;
      if (key1 > qr || key1 >= T_) s1 = -3.0e38f;
      float mx = fmaxf(s0, s1);
#pragma unroll
      for (int o = 1; o < 16; o <<= 1) mx = fmaxf(mx, __shfl_xor(mx, o, 32));
      const float mn = fmaxf(m_[r], mx);
      const float p0 = __expf(s0 - mn);
      const float p1 = __expf(s1 - mn);
      float rs = p0 + p1;
#pragma unroll
      for (int o = 1; o < 16; o <<= 1) rs += __shfl_xor(rs, o, 32);
      const float sf = __expf(m_[r] - mn);
      l_[r] = l_[r] * sf + rs;
      m_[r] = mn;
      sf_[r] = sf;
      Pw[(r + hi*8)*32 + l16]      = (__bf16)p0;  // S-layout -> A-layout
      Pw[(r + hi*8)*32 + 16 + l16] = (__bf16)p1;
    }
    const v16bf pf = ld_frag(&Pw[l16*32 + hi*8]);
#pragma unroll
    for (int n = 0; n < 4; ++n) {
#pragma unroll
      for (int r = 0; r < 8; ++r) acc[n][r] *= sf_[r];
      v16bf vf = ld_frag(&Vl[(n*16 + l16)*VST + kb + hi*8]);
      acc[n] = wmma_bf16(pf, vf, acc[n]);
    }
  }

#pragma unroll
  for (int r = 0; r < 8; ++r) {
    const int q = q0 + r + hi*8;
    if (q < T_) {
      const float inv = 1.0f / l_[r];
#pragma unroll
      for (int n = 0; n < 4; ++n)
        O[base + (size_t)q * C_ + n*16 + l16] = (__bf16)(acc[n][r] * inv);
    }
  }
}

// ---------------------------------------------------------------------------
extern "C" void kernel_launch(void* const* d_in, const int* in_sizes, int n_in,
                              void* d_out, int out_size, void* d_ws, size_t ws_size,
                              hipStream_t stream) {
  const float* X  = (const float*)d_in[0];
  const float* Wq = (const float*)d_in[1];
  const float* bq = (const float*)d_in[2];
  const float* Wk = (const float*)d_in[3];
  const float* bk = (const float*)d_in[4];
  const float* Wv = (const float*)d_in[5];
  const float* bv = (const float*)d_in[6];
  const float* Wo = (const float*)d_in[7];
  const float* bo = (const float*)d_in[8];
  float* out = (float*)d_out;

  __bf16* ws = (__bf16*)d_ws;
  const size_t S = (size_t)M_ * C_;
  __bf16* Qb  = ws;
  __bf16* Kb  = ws + S;
  __bf16* Vb  = ws + 2*S;
  __bf16* Cb  = ws + 3*S;
  __bf16* Xb  = ws + 4*S;
  __bf16* Wtq = ws + 5*S;
  __bf16* Wtk = Wtq + W2_;
  __bf16* Wtv = Wtk + W2_;
  __bf16* Wto = Wtv + W2_;

  dim3 bb(256);
  // Prep: X -> bf16 (once), weights -> bf16 transposed [n][k]
  cvt_kernel<<<dim3((M_ * C_ / 4) / 256), bb, 0, stream>>>(X, Xb);
  wtr_kernel<<<dim3(W2_/256), bb, 0, stream>>>(Wq, Wtq);
  wtr_kernel<<<dim3(W2_/256), bb, 0, stream>>>(Wk, Wtk);
  wtr_kernel<<<dim3(W2_/256), bb, 0, stream>>>(Wv, Wtv);
  wtr_kernel<<<dim3(W2_/256), bb, 0, stream>>>(Wo, Wto);

  dim3 gg(M_/128, C_/128);   // 394 x 6
  gemm_kernel<true><<<gg, bb, 0, stream>>>(Xb, Wtq, bq, Qb);
  gemm_kernel<true><<<gg, bb, 0, stream>>>(Xb, Wtk, bk, Kb);
  gemm_kernel<true><<<gg, bb, 0, stream>>>(Xb, Wtv, bv, Vb);

  dim3 gf(2, H_, B_);        // 2 query slabs x 12 heads x 256 batch
  flash_kernel<<<gf, bb, 0, stream>>>(Qb, Kb, Vb, Cb);

  gemm_kernel<false><<<gg, bb, 0, stream>>>(Cb, Wto, bo, out);
}
